// MultiHeadAttention_1589137900017
// MI455X (gfx1250) — matrix operand order repaired and checked by reading, not compiled, not run on hardware
//
#include <hip/hip_runtime.h>

typedef __attribute__((ext_vector_type(16))) __bf16 v16bf;
typedef __attribute__((ext_vector_type(8)))  __bf16 v8bf;
typedef __attribute__((ext_vector_type(8)))  float  v8f;

constexpr int BATCH = 2;
constexpr int SEQ   = 2048;
constexpr int DM    = 1024;
constexpr int H     = 16;
constexpr int DK    = 64;
constexpr int ROWS  = BATCH * SEQ;   // 4096 flattened tokens

// ---------------------------------------------------------------------------
// Tile loaders matching CDNA5 WMMA VGPR layouts (wave32).
// A matrix 16x32 bf16: lane L -> row M = L%16; elems j<8 : K = half*8+j
//                                       j>=8: K = 16 + half*8 + (j-8)
// => two contiguous 16B loads at col offsets half*8 and half*8+16.
// ---------------------------------------------------------------------------
__device__ inline v16bf load_tileA(const __bf16* base, int ld) {
  const int lane = threadIdx.x & 31;
  const int row  = lane & 15;
  const int half = lane >> 4;
  const __bf16* p = base + (long)row * ld + half * 8;
  v8bf lo = *(const v8bf*)(p);
  v8bf hi = *(const v8bf*)(p + 16);
  v16bf r;
#pragma unroll
  for (int i = 0; i < 8; ++i) { r[i] = lo[i]; r[i + 8] = hi[i]; }
  return r;
}

// B matrix 32x16 bf16: lane L -> col N = L%16; elems j<8 : K = half*8+j
//                                       j>=8: K = 16 + half*8 + (j-8)
// => two contiguous 16B loads at col offsets half*8 and half*8+16.
__device__ inline v16bf load_tileB(const __bf16* base, int ld) {
  const int lane = threadIdx.x & 31;
  const int row  = lane & 15;          // N
  const int half = lane >> 4;
  const __bf16* p = base + (long)row * ld + half * 8;
  v8bf lo = *(const v8bf*)(p);
  v8bf hi = *(const v8bf*)(p + 16);
  v16bf r;
#pragma unroll
  for (int i = 0; i < 8; ++i) { r[i] = lo[i]; r[i + 8] = hi[i]; }
  return r;
}

__device__ inline v8f wmma_bf16(v16bf a, v16bf b, v8f c) {
  return __builtin_amdgcn_wmma_f32_16x16x32_bf16(false, a, false, b,
                                                 (short)0, c, false, false);
}

// global_prefetch_b8 of this lane's row at the given base (gfx1250 path).
__device__ inline void prefetch_row(const __bf16* base, int ld) {
  const int row = threadIdx.x & 15;
  __builtin_prefetch(base + (long)row * ld, 0, 0);
}

// ---------------------------------------------------------------------------
// 1) fp32 -> bf16 conversion
// ---------------------------------------------------------------------------
__global__ void cvt_bf16(const float* __restrict__ in, __bf16* __restrict__ out, int n) {
  int i = blockIdx.x * blockDim.x + threadIdx.x;
  int stride = gridDim.x * blockDim.x;
  for (; i < n; i += stride) out[i] = (__bf16)in[i];
}

// ---------------------------------------------------------------------------
// 2) Q/K/V projection (y = x @ W^T) + RoPE on Q,K. One wave -> 32 rows x 64
//    cols (one full head): each B (weight) tile feeds two row sub-tiles.
//    V stored transposed [bh][dk][s].
// ---------------------------------------------------------------------------
__global__ void qkv_rope_kernel(const __bf16* __restrict__ xb,
                                const __bf16* __restrict__ Wq,
                                const __bf16* __restrict__ Wk,
                                const __bf16* __restrict__ Wv,
                                __bf16* __restrict__ Qr,
                                __bf16* __restrict__ Kr,
                                __bf16* __restrict__ Vt) {
  int id   = blockIdx.x;
  int mat  = id % 3; id /= 3;           // 0=Q 1=K 2=V
  int head = id % H; id /= H;
  int rt   = id;                        // row tile 0..127 (32 rows each)
  const int lane = threadIdx.x & 31;
  const int half = lane >> 4, ln = lane & 15;

  const __bf16* W = (mat == 0) ? Wq : (mat == 1) ? Wk : Wv;
  const int row0 = rt * 32;
  const __bf16* aBase0 = xb + (long)row0 * DM;
  const __bf16* aBase1 = xb + (long)(row0 + 16) * DM;
  const __bf16* wBase  = W + (long)(head * DK) * DM;

  v8f acc[2][4] = {};
#pragma unroll 1
  for (int k0 = 0; k0 < DM; k0 += 32) {
    v16bf a0 = load_tileA(aBase0 + k0, DM);
    v16bf a1 = load_tileA(aBase1 + k0, DM);
    if (k0 + 128 < DM) {                 // pull lines 4 K-steps ahead
      prefetch_row(aBase0 + k0 + 128, DM);
      prefetch_row(aBase1 + k0 + 128, DM);
      prefetch_row(wBase + k0 + 128, DM);
      prefetch_row(wBase + 32 * DM + k0 + 128, DM);
    }
#pragma unroll
    for (int t = 0; t < 4; ++t) {
      v16bf b = load_tileB(wBase + (long)(t * 16) * DM + k0, DM);
      acc[0][t] = wmma_bf16(a0, b, acc[0][t]);
      acc[1][t] = wmma_bf16(a1, b, acc[1][t]);
    }
  }

  const int b_idx = row0 / SEQ;
  const int s0    = row0 % SEQ;
  const int bh    = b_idx * H + head;

  if (mat < 2) {
    __bf16* dst = (mat == 0) ? Qr : Kr;
#pragma unroll
    for (int u = 0; u < 2; ++u) {
      // RoPE: tiles (0,2) and (1,3) hold partners dq and dq+32.
#pragma unroll
      for (int t = 0; t < 2; ++t) {
        const int tpos = t * 16 + ln;                    // dq % 32
        // theta = 10000^(-tpos/32) = exp(-tpos * ln(10000)/32)
        const float theta = __expf(-(float)tpos * 0.28782313662425583f);
#pragma unroll
        for (int r = 0; r < 8; ++r) {
          int m = s0 + u * 16 + r + 8 * half;
          float ang = (float)m * theta;
          float cs = __cosf(ang), sn = __sinf(ang);
          float lo = acc[u][t][r], hi = acc[u][t + 2][r];
          acc[u][t][r]     = lo * cs - hi * sn;
          acc[u][t + 2][r] = hi * cs + lo * sn;
        }
      }
#pragma unroll
      for (int t = 0; t < 4; ++t)
#pragma unroll
        for (int r = 0; r < 8; ++r) {
          int s  = s0 + u * 16 + r + 8 * half;
          int dk = t * 16 + ln;
          dst[((long)bh * SEQ + s) * DK + dk] = (__bf16)acc[u][t][r];
        }
    }
  } else {
#pragma unroll
    for (int u = 0; u < 2; ++u)
#pragma unroll
      for (int t = 0; t < 4; ++t)
#pragma unroll
        for (int r = 0; r < 8; ++r) {
          int s  = s0 + u * 16 + r + 8 * half;
          int dk = t * 16 + ln;
          Vt[((long)bh * DK + dk) * SEQ + s] = (__bf16)acc[u][t][r];
        }
  }
}

// ---------------------------------------------------------------------------
// 3) Causal flash attention. One wave per (b,h,32-query tile); 32-key chunks.
//    Each K/V tile load is shared by both 16-query sub-tiles.
// ---------------------------------------------------------------------------
__global__ void flash_attn_kernel(const __bf16* __restrict__ Qr,
                                  const __bf16* __restrict__ Kr,
                                  const __bf16* __restrict__ Vt,
                                  __bf16* __restrict__ O) {
  __shared__ alignas(16) __bf16 ldsP[32 * 32];   // P staging (C->A relayout)

  int id = blockIdx.x;
  int qt = id % (SEQ / 32); id /= (SEQ / 32);
  int bh = id;
  int b  = bh / H, h = bh % H;
  const int lane = threadIdx.x & 31;
  const int half = lane >> 4, ln = lane & 15;
  const int q0 = qt * 32;

  const __bf16* Qb = Qr + (long)bh * SEQ * DK;
  const __bf16* Kb = Kr + (long)bh * SEQ * DK;
  const __bf16* Vb = Vt + (long)bh * DK * SEQ;

  // Q A-tiles: [row sub-tile u][dk half], reused across all key chunks.
  v16bf aQ[2][2];
#pragma unroll
  for (int u = 0; u < 2; ++u) {
    aQ[u][0] = load_tileA(Qb + (long)(q0 + u * 16) * DK + 0, DK);
    aQ[u][1] = load_tileA(Qb + (long)(q0 + u * 16) * DK + 32, DK);
  }

  v8f oacc[2][4] = {};
  float mrun[2][8], lrun[2][8];
#pragma unroll
  for (int u = 0; u < 2; ++u)
#pragma unroll
    for (int r = 0; r < 8; ++r) { mrun[u][r] = -__builtin_inff(); lrun[u][r] = 0.f; }

  const float scale = 0.125f;       // 1/sqrt(64)
  const int jend = q0 + 32;         // causal: need keys < jend

  for (int j0 = 0; j0 < jend; j0 += 32) {
    // ---- scores: s[u][t], keys (j0 + t*16 .. +15) vs queries sub-tile u ----
    v8f s[2][2] = {};
#pragma unroll
    for (int t = 0; t < 2; ++t) {
      v16bf bk0 = load_tileB(Kb + (long)(j0 + t * 16) * DK + 0, DK);
      v16bf bk1 = load_tileB(Kb + (long)(j0 + t * 16) * DK + 32, DK);
#pragma unroll
      for (int u = 0; u < 2; ++u) {
        s[u][t] = wmma_bf16(aQ[u][0], bk0, s[u][t]);
        s[u][t] = wmma_bf16(aQ[u][1], bk1, s[u][t]);
      }
    }
    // ---- scale + causal mask + online softmax ----
#pragma unroll
    for (int u = 0; u < 2; ++u) {
      const int key0 = j0 + ln, key1 = j0 + 16 + ln;
#pragma unroll
      for (int r = 0; r < 8; ++r) {
        int q = q0 + u * 16 + r + 8 * half;
        float v0 = s[u][0][r] * scale, v1 = s[u][1][r] * scale;
        s[u][0][r] = (key0 <= q) ? v0 : -__builtin_inff();
        s[u][1][r] = (key1 <= q) ? v1 : -__builtin_inff();
      }
#pragma unroll
      for (int r = 0; r < 8; ++r) {
        float mx = fmaxf(s[u][0][r], s[u][1][r]);
#pragma unroll
        for (int d = 1; d < 16; d <<= 1) mx = fmaxf(mx, __shfl_xor(mx, d, 32));
        float mnew = fmaxf(mrun[u][r], mx);
        float corr = __expf(mrun[u][r] - mnew);
        float p0 = __expf(s[u][0][r] - mnew);
        float p1 = __expf(s[u][1][r] - mnew);
        s[u][0][r] = p0; s[u][1][r] = p1;
        float rs = p0 + p1;
#pragma unroll
        for (int d = 1; d < 16; d <<= 1) rs += __shfl_xor(rs, d, 32);
        lrun[u][r] = lrun[u][r] * corr + rs;
        mrun[u][r] = mnew;
#pragma unroll
        for (int t = 0; t < 4; ++t) oacc[u][t][r] *= corr;
      }
      // ---- P: C-layout f32 -> row-major bf16 in LDS ----
#pragma unroll
      for (int r = 0; r < 8; ++r) {
        int m = u * 16 + r + 8 * half;
        ldsP[m * 32 + ln]      = (__bf16)s[u][0][r];
        ldsP[m * 32 + 16 + ln] = (__bf16)s[u][1][r];
      }
    }
    __syncthreads();
    v16bf aP0 = load_tileA(ldsP, 32);
    v16bf aP1 = load_tileA(ldsP + 16 * 32, 32);
    __syncthreads();
    // ---- o += P @ V  (Vt B tile shared by both query sub-tiles) ----
#pragma unroll
    for (int t = 0; t < 4; ++t) {
      v16bf bv = load_tileB(Vb + (long)(t * 16) * SEQ + j0, SEQ);
      oacc[0][t] = wmma_bf16(aP0, bv, oacc[0][t]);
      oacc[1][t] = wmma_bf16(aP1, bv, oacc[1][t]);
    }
  }

  // ---- normalize + store concat-head output [b][s][d_model] ----
#pragma unroll
  for (int u = 0; u < 2; ++u)
#pragma unroll
    for (int r = 0; r < 8; ++r) {
      float inv = 1.0f / lrun[u][r];
      int s_row = q0 + u * 16 + r + 8 * half;
#pragma unroll
      for (int t = 0; t < 4; ++t) {
        float v = oacc[u][t][r] * inv;
        O[((long)b * SEQ + s_row) * DM + h * DK + t * 16 + ln] = (__bf16)v;
      }
    }
}

// ---------------------------------------------------------------------------
// 4) Output projection: out = O @ W_O^T, 32x64 register tile, fp32 stores.
// ---------------------------------------------------------------------------
__global__ void out_proj_kernel(const __bf16* __restrict__ Ob,
                                const __bf16* __restrict__ Wo,
                                float* __restrict__ out) {
  int id = blockIdx.x;
  int ct = id % (DM / 64); id /= (DM / 64);
  int rt = id;                          // 0..127
  const int lane = threadIdx.x & 31;
  const int half = lane >> 4, ln = lane & 15;
  const int row0 = rt * 32;
  const __bf16* aBase0 = Ob + (long)row0 * DM;
  const __bf16* aBase1 = Ob + (long)(row0 + 16) * DM;
  const __bf16* wBase  = Wo + (long)(ct * 64) * DM;

  v8f acc[2][4] = {};
#pragma unroll 1
  for (int k0 = 0; k0 < DM; k0 += 32) {
    v16bf a0 = load_tileA(aBase0 + k0, DM);
    v16bf a1 = load_tileA(aBase1 + k0, DM);
    if (k0 + 128 < DM) {
      prefetch_row(aBase0 + k0 + 128, DM);
      prefetch_row(aBase1 + k0 + 128, DM);
      prefetch_row(wBase + k0 + 128, DM);
      prefetch_row(wBase + 32 * DM + k0 + 128, DM);
    }
#pragma unroll
    for (int t = 0; t < 4; ++t) {
      v16bf b = load_tileB(wBase + (long)(t * 16) * DM + k0, DM);
      acc[0][t] = wmma_bf16(a0, b, acc[0][t]);
      acc[1][t] = wmma_bf16(a1, b, acc[1][t]);
    }
  }
#pragma unroll
  for (int u = 0; u < 2; ++u)
#pragma unroll
    for (int t = 0; t < 4; ++t)
#pragma unroll
      for (int r = 0; r < 8; ++r)
        out[(long)(row0 + u * 16 + r + 8 * half) * DM + ct * 64 + t * 16 + ln]
            = acc[u][t][r];
}

// ---------------------------------------------------------------------------
extern "C" void kernel_launch(void* const* d_in, const int* in_sizes, int n_in,
                              void* d_out, int out_size, void* d_ws, size_t ws_size,
                              hipStream_t stream) {
  (void)in_sizes; (void)n_in; (void)out_size; (void)ws_size;
  const float* x  = (const float*)d_in[0];
  const float* Wq = (const float*)d_in[1];
  const float* Wk = (const float*)d_in[2];
  const float* Wv = (const float*)d_in[3];
  const float* Wo = (const float*)d_in[4];

  char* base = (char*)d_ws;
  size_t off = 0;
  auto take = [&](size_t bytes) -> void* {
    void* p = base + off;
    off += (bytes + 255) & ~(size_t)255;
    return p;
  };
  __bf16* xb  = (__bf16*)take((size_t)ROWS * DM * 2);            // 8 MB
  __bf16* wqb = (__bf16*)take((size_t)DM * DM * 2);              // 2 MB
  __bf16* wkb = (__bf16*)take((size_t)DM * DM * 2);
  __bf16* wvb = (__bf16*)take((size_t)DM * DM * 2);
  __bf16* wob = (__bf16*)take((size_t)DM * DM * 2);
  __bf16* Qr  = (__bf16*)take((size_t)BATCH * H * SEQ * DK * 2); // 8 MB
  __bf16* Kr  = (__bf16*)take((size_t)BATCH * H * SEQ * DK * 2);
  __bf16* Vt  = (__bf16*)take((size_t)BATCH * H * DK * SEQ * 2);
  __bf16* Ob  = (__bf16*)take((size_t)ROWS * DM * 2);            // 8 MB

  cvt_bf16<<<4096, 256, 0, stream>>>(x,  xb,  ROWS * DM);
  cvt_bf16<<<1024, 256, 0, stream>>>(Wq, wqb, DM * DM);
  cvt_bf16<<<1024, 256, 0, stream>>>(Wk, wkb, DM * DM);
  cvt_bf16<<<1024, 256, 0, stream>>>(Wv, wvb, DM * DM);
  cvt_bf16<<<1024, 256, 0, stream>>>(Wo, wob, DM * DM);

  qkv_rope_kernel<<<(ROWS / 32) * H * 3, 32, 0, stream>>>(xb, wqb, wkb, wvb, Qr, Kr, Vt);
  flash_attn_kernel<<<BATCH * H * (SEQ / 32), 32, 0, stream>>>(Qr, Kr, Vt, Ob);
  out_proj_kernel<<<(ROWS / 32) * (DM / 64), 32, 0, stream>>>(Ob, wob, (float*)d_out);
}
